// UnifiedDecoder_66855460930110
// MI455X (gfx1250) — matrix operand order repaired and checked by reading, not compiled, not run on hardware
//
#include <hip/hip_runtime.h>
#include <stdint.h>

// ---------------- CDNA5 WMMA plumbing ----------------
typedef __attribute__((ext_vector_type(16))) __bf16 bf16x16;
typedef __attribute__((ext_vector_type(8)))  float   f32x8;
typedef unsigned int u32x4 __attribute__((ext_vector_type(4)));
typedef int          i32x8 __attribute__((ext_vector_type(8)));
typedef int          i32x4 __attribute__((ext_vector_type(4)));

#define HD   512
#define LSEQ 64
#define NTOT 2048

#define NEWH_OFF 0
#define HPN_OFF  1048576
#define NUC_OFF  1056768
#define STOP_OFF 1712128

#if defined(__has_builtin)
#if __has_builtin(__builtin_amdgcn_tensor_load_to_lds)
#define HAVE_TDM 1
#endif
#endif

__device__ __forceinline__ f32x8 fzero8() {
  f32x8 z = {0.f,0.f,0.f,0.f,0.f,0.f,0.f,0.f};
  return z;
}
__device__ __forceinline__ float sigm(float x) { return 1.0f / (1.0f + __expf(-x)); }

// A fragment (16x32 bf16): lane L holds row m=L&15; half=L>>4.
// chunks at k0+half*8 (elems 0..7) and k0+16+half*8 (elems 8..15)
__device__ __forceinline__ bf16x16 frag_a(const __bf16* p) {
  union { bf16x16 v; float4 q[2]; } u;
  u.q[0] = *reinterpret_cast<const float4*>(p);
  u.q[1] = *reinterpret_cast<const float4*>(p + 16);
  return u.v;
}
// B fragment (32x16 bf16): lane L holds col n=L&15; half=L>>4;
// chunks at W[n][k0+half*8] (elems 0..7) and W[n][k0+16+half*8] (elems 8..15)
__device__ __forceinline__ bf16x16 frag_b(const __bf16* p) {
  union { bf16x16 v; float4 q[2]; } u;
  u.q[0] = *reinterpret_cast<const float4*>(p);
  u.q[1] = *reinterpret_cast<const float4*>(p + 16);
  return u.v;
}
__device__ __forceinline__ f32x8 wmma_bf16(bf16x16 a, bf16x16 b, f32x8 c) {
  return __builtin_amdgcn_wmma_f32_16x16x32_bf16(false, a, false, b, (short)0, c, false, false);
}

// ---------------- prep kernels ----------------
__global__ void cvt_slice(const float* __restrict__ src, __bf16* __restrict__ dst,
                          int rows, int cols, int sstride, int soff, int trows) {
  int i = blockIdx.x * 256 + threadIdx.x;
  int total = trows * cols;
  if (i >= total) return;
  int r = i / cols, c = i % cols;
  float v = (r < rows) ? src[(size_t)r * sstride + soff + c] : 0.0f;
  dst[i] = (__bf16)v;
}

__global__ void copy_slice(const float* __restrict__ src, float* __restrict__ dst,
                           int rows, int cols, int sstride, int soff) {
  int i = blockIdx.x * 256 + threadIdx.x;
  if (i >= rows * cols) return;
  int r = i / cols, c = i % cols;
  dst[i] = src[(size_t)r * sstride + soff + c];
}

// ---------------- lat_part[n][o] = b_nuc_nl[o] + glat[n,:] @ W_nuc_nl[o, 512:640] ----------------
__global__ __launch_bounds__(256) void lat_kernel(const __bf16* __restrict__ glatb,
                                                  const __bf16* __restrict__ Wlatb,
                                                  const float* __restrict__ b_nl,
                                                  float* __restrict__ lat) {
  const int row0 = blockIdx.x * 16;
  const int tid = threadIdx.x;
  const int wv = tid >> 5, lane = tid & 31;
  const int lh = lane >> 4, lc = lane & 15;
  const int cb = wv * 64;

  f32x8 ac[4];
#pragma unroll
  for (int c = 0; c < 4; ++c) ac[c] = fzero8();
  for (int ks = 0; ks < 4; ++ks) {
    bf16x16 a = frag_a(&glatb[(size_t)(row0 + lc) * 128 + ks * 32 + lh * 8]);
#pragma unroll
    for (int c = 0; c < 4; ++c) {
      int o = cb + c * 16 + lc;
      ac[c] = wmma_bf16(a, frag_b(&Wlatb[(size_t)o * 128 + ks * 32 + lh * 8]), ac[c]);
    }
  }
#pragma unroll
  for (int c = 0; c < 4; ++c) {
    int o = cb + c * 16 + lc;
#pragma unroll
    for (int i = 0; i < 8; ++i) {
      int m = i + 8 * lh;
      lat[(size_t)(row0 + m) * HD + o] = ac[c][i] + b_nl[o];
    }
  }
}

// ---------------- GraphGRU message kernel (+ stop/topo head) ----------------
__global__ __launch_bounds__(256) void msg_kernel(
    const int* __restrict__ hpn_idx, const float* __restrict__ h_nei,
    const __bf16* __restrict__ Wzb, const float* __restrict__ Wzsel, const float* __restrict__ bz,
    const float* __restrict__ Wr_mp, const float* __restrict__ br,
    const __bf16* __restrict__ Urb,
    const __bf16* __restrict__ Whb, const float* __restrict__ Whsel, const float* __restrict__ bh,
    const __bf16* __restrict__ Wtopolb, const float* __restrict__ b_topo_nl,
    const __bf16* __restrict__ Wtopob, const float* __restrict__ b_topo,
    float* __restrict__ msg, float* __restrict__ out) {
  extern __shared__ __align__(16) char smem[];
  __bf16* hn  = (__bf16*)smem;                              // [128][512] bf16, 128 KB
  float*  shf = (float*)(smem + 131072);                    // [16][512] f32,  32 KB
  __bf16* shb = (__bf16*)(smem + 131072 + 32768);           // [16][512] bf16, 16 KB
  __bf16* sgb = (__bf16*)(smem + 131072 + 32768 + 16384);   // [16][512] bf16, 16 KB
  __bf16* mgb = (__bf16*)(smem + 131072 + 32768 + 32768);   // [16][512] bf16, 16 KB
  __bf16* hid = hn;                                         // reuse (hn dead after r-phase)

  const int row0 = blockIdx.x * 16;
  const int tid = threadIdx.x;
  const int wv = tid >> 5, lane = tid & 31;
  const int lh = lane >> 4, lc = lane & 15;
  const int cb = wv * 64;

  const float* src = h_nei + (size_t)row0 * (8 * HD);
  for (int i4 = tid; i4 < (16 * 8 * HD) / 4; i4 += 256) {
    float4 v = reinterpret_cast<const float4*>(src)[i4];
    hn[i4 * 4 + 0] = (__bf16)v.x;
    hn[i4 * 4 + 1] = (__bf16)v.y;
    hn[i4 * 4 + 2] = (__bf16)v.z;
    hn[i4 * 4 + 3] = (__bf16)v.w;
  }
  for (int j = tid; j < 16 * HD; j += 256) {
    int n = j >> 9, o = j & 511;
    float s = 0.f;
#pragma unroll
    for (int k = 0; k < 8; ++k) s += src[(size_t)(n * 8 + k) * HD + o];
    shf[j] = s;
    shb[j] = (__bf16)s;
  }
  __syncthreads();

  // r gate + gated neighbor sum (lane's 8 accumulator elements == 8 neighbors of one node)
  for (int rt = 0; rt < 8; ++rt) {
    f32x8 rac[4];
#pragma unroll
    for (int c = 0; c < 4; ++c) rac[c] = fzero8();
    for (int ks = 0; ks < 16; ++ks) {
      bf16x16 a = frag_a(&hn[(size_t)(rt * 16 + lc) * HD + ks * 32 + lh * 8]);
#pragma unroll
      for (int c = 0; c < 4; ++c) {
        int o = cb + c * 16 + lc;
        rac[c] = wmma_bf16(a, frag_b(&Urb[(size_t)o * HD + ks * 32 + lh * 8]), rac[c]);
      }
    }
    int n = 2 * rt + lh;
    int hp = hpn_idx[row0 + n];
#pragma unroll
    for (int c = 0; c < 4; ++c) {
      int o = cb + c * 16 + lc;
      float sel = Wr_mp[o * 4 + hp] + br[o];
      float s = 0.f;
#pragma unroll
      for (int i = 0; i < 8; ++i) {
        float r = sigm(rac[c][i] + sel);
        float hv = (float)hn[(size_t)(rt * 16 + lh * 8 + i) * HD + o];
        s += r * hv;
      }
      sgb[n * HD + o] = (__bf16)s;
    }
  }
  __syncthreads();

  f32x8 zac[4], pac[4];
#pragma unroll
  for (int c = 0; c < 4; ++c) { zac[c] = fzero8(); pac[c] = fzero8(); }
  for (int ks = 0; ks < 16; ++ks) {
    bf16x16 az = frag_a(&shb[(size_t)lc * HD + ks * 32 + lh * 8]);
    bf16x16 ag = frag_a(&sgb[(size_t)lc * HD + ks * 32 + lh * 8]);
#pragma unroll
    for (int c = 0; c < 4; ++c) {
      int o = cb + c * 16 + lc;
      zac[c] = wmma_bf16(az, frag_b(&Wzb[(size_t)o * HD + ks * 32 + lh * 8]), zac[c]);
      pac[c] = wmma_bf16(ag, frag_b(&Whb[(size_t)o * HD + ks * 32 + lh * 8]), pac[c]);
    }
  }
  int hpm[8];
#pragma unroll
  for (int i = 0; i < 8; ++i) hpm[i] = hpn_idx[row0 + i + 8 * lh];
#pragma unroll
  for (int c = 0; c < 4; ++c) {
    int o = cb + c * 16 + lc;
#pragma unroll
    for (int i = 0; i < 8; ++i) {
      int m = i + 8 * lh;
      float z   = sigm(zac[c][i] + Wzsel[o * 4 + hpm[i]] + bz[o]);
      float pre = tanhf(pac[c][i] + Whsel[o * 4 + hpm[i]] + bh[o]);
      float sh  = shf[m * HD + o];
      float mg  = (1.0f - z) * sh + z * pre;
      msg[(size_t)(row0 + m) * HD + o] = mg;
      mgb[m * HD + o] = (__bf16)mg;
    }
  }
  __syncthreads();

  f32x8 tac[4];
#pragma unroll
  for (int c = 0; c < 4; ++c) tac[c] = fzero8();
  for (int ks = 0; ks < 16; ++ks) {
    bf16x16 a = frag_a(&mgb[(size_t)lc * HD + ks * 32 + lh * 8]);
#pragma unroll
    for (int c = 0; c < 4; ++c) {
      int o = cb + c * 16 + lc;
      tac[c] = wmma_bf16(a, frag_b(&Wtopolb[(size_t)o * HD + ks * 32 + lh * 8]), tac[c]);
    }
  }
#pragma unroll
  for (int c = 0; c < 4; ++c) {
    int o = cb + c * 16 + lc;
#pragma unroll
    for (int i = 0; i < 8; ++i) {
      int m = i + 8 * lh;
      float v = tac[c][i] + b_topo_nl[o];
      hid[m * HD + o] = (__bf16)(v > 0.f ? v : 0.f);
    }
  }
  __syncthreads();
  if (wv == 0) {
    f32x8 cfr = fzero8();
    for (int ks = 0; ks < 16; ++ks)
      cfr = wmma_bf16(frag_a(&hid[(size_t)lc * HD + ks * 32 + lh * 8]),
                      frag_b(&Wtopob[(size_t)lc * HD + ks * 32 + lh * 8]), cfr);
    if (lc == 0) {
#pragma unroll
      for (int i = 0; i < 8; ++i) out[STOP_OFF + row0 + i + 8 * lh] = cfr[i] + b_topo[0];
    }
  }
}

// ---------------- 64-step GRU scan, 32 rows/block, fused heads, TDM-staged lat ----------------
__global__ __launch_bounds__(256) void scan_kernel(
    const int* __restrict__ nuc_idx, const float* __restrict__ msg, const float* __restrict__ lat,
    const __bf16* __restrict__ Wznb, const __bf16* __restrict__ Wrnb, const __bf16* __restrict__ Whnb,
    const float* __restrict__ Wznsel, const float* __restrict__ Wrnsel, const float* __restrict__ Whnsel,
    const float* __restrict__ bz, const float* __restrict__ br, const float* __restrict__ bh,
    const __bf16* __restrict__ Wnlb, const __bf16* __restrict__ Wnucb, const float* __restrict__ b_nuc,
    const __bf16* __restrict__ Whpnlb, const float* __restrict__ b_hpn_nl,
    const __bf16* __restrict__ Whpnb, const float* __restrict__ b_hpn,
    float* __restrict__ out) {
  extern __shared__ __align__(16) char smem2[];
  __bf16* h_tile   = (__bf16*)smem2;                 // [32][512] bf16, 32 KB
  __bf16* rh_tile  = (__bf16*)(smem2 + 32768);       // [32][512] bf16, 32 KB
  __bf16* hid_tile = (__bf16*)(smem2 + 65536);       // [32][512] bf16, 32 KB
  float*  lat_lds  = (float*)(smem2 + 98304);        // [32][512] f32,  64 KB

  const int row0 = blockIdx.x * 32;
  const int tid = threadIdx.x;
  const int wv = tid >> 5, lane = tid & 31;
  const int lh = lane >> 4, lc = lane & 15;
  const int cb = wv * 64;

  // ---- stage lat slice (reused 64x) into LDS via Tensor Data Mover ----
#if HAVE_TDM
  if (wv == 0) {
    unsigned lds_off = (unsigned)(size_t)lat_lds;
    unsigned long long ga = (unsigned long long)(size_t)(lat + (size_t)row0 * HD);
    u32x4 g0;
    g0[0] = 1u;                                              // count=1, user descriptor
    g0[1] = lds_off;                                         // lds_addr
    g0[2] = (unsigned)ga;                                    // global_addr[31:0]
    g0[3] = (unsigned)((ga >> 32) & 0x01FFFFFFu) | (2u << 30); // global_addr[56:32], type=2
    i32x8 g1;
    g1[0] = 0x00020000;          // workgroup_mask=0, data_size=2 (4B)
    g1[1] = (int)(16384u << 16); // tensor_dim0 = 16384 elements (low 16 in [31:16])
    g1[2] = (int)(1u << 16);     // tensor_dim0 hi=0, tensor_dim1=1 (low 16)
    g1[3] = (int)(16384u << 16); // tensor_dim1 hi=0, tile_dim0 = 16384
    g1[4] = 0;                   // tile_dim1=0 (unused), tile_dim2=0
    g1[5] = 16384;               // tensor_dim0_stride low 32
    g1[6] = 0;                   // stride hi / dim1_stride
    g1[7] = 0;
    i32x4 gz; gz[0] = 0; gz[1] = 0; gz[2] = 0; gz[3] = 0;
#if __clang_major__ >= 23
    i32x8 g5; g5[0]=0; g5[1]=0; g5[2]=0; g5[3]=0; g5[4]=0; g5[5]=0; g5[6]=0; g5[7]=0;
    __builtin_amdgcn_tensor_load_to_lds(g0, g1, gz, gz, g5, 0);
#else
    __builtin_amdgcn_tensor_load_to_lds(g0, g1, gz, gz, 0);
#endif
    __builtin_amdgcn_s_wait_tensorcnt(0);
  }
#else
  for (int i = tid; i < 32 * HD; i += 256) lat_lds[i] = lat[(size_t)row0 * HD + i];
#endif

  for (int i = tid; i < 32 * HD; i += 256) h_tile[i] = (__bf16)msg[(size_t)row0 * HD + i];
  __syncthreads();

  for (int t = 0; t < LSEQ; ++t) {
    int xidx[2][8];
#pragma unroll
    for (int rt = 0; rt < 2; ++rt)
#pragma unroll
      for (int i = 0; i < 8; ++i)
        xidx[rt][i] = nuc_idx[(row0 + rt * 16 + i + 8 * lh) * LSEQ + t];

    // ---- z/r gates: each B fragment feeds 2 row-tiles ----
    f32x8 zac[2][4], rac[2][4];
#pragma unroll
    for (int rt = 0; rt < 2; ++rt)
#pragma unroll
      for (int c = 0; c < 4; ++c) { zac[rt][c] = fzero8(); rac[rt][c] = fzero8(); }
    for (int ks = 0; ks < 16; ++ks) {
      bf16x16 a0 = frag_a(&h_tile[(size_t)lc * HD + ks * 32 + lh * 8]);
      bf16x16 a1 = frag_a(&h_tile[(size_t)(16 + lc) * HD + ks * 32 + lh * 8]);
#pragma unroll
      for (int c = 0; c < 4; ++c) {
        int o = cb + c * 16 + lc;
        bf16x16 bzf = frag_b(&Wznb[(size_t)o * HD + ks * 32 + lh * 8]);
        zac[0][c] = wmma_bf16(a0, bzf, zac[0][c]);
        zac[1][c] = wmma_bf16(a1, bzf, zac[1][c]);
        bf16x16 brf = frag_b(&Wrnb[(size_t)o * HD + ks * 32 + lh * 8]);
        rac[0][c] = wmma_bf16(a0, brf, rac[0][c]);
        rac[1][c] = wmma_bf16(a1, brf, rac[1][c]);
      }
    }
#pragma unroll
    for (int rt = 0; rt < 2; ++rt)
#pragma unroll
      for (int c = 0; c < 4; ++c) {
        int o = cb + c * 16 + lc;
#pragma unroll
        for (int i = 0; i < 8; ++i) {
          int m = rt * 16 + i + 8 * lh;
          float z = sigm(zac[rt][c][i] + Wznsel[o * 5 + xidx[rt][i]] + bz[o]);
          float r = sigm(rac[rt][c][i] + Wrnsel[o * 5 + xidx[rt][i]] + br[o]);
          zac[rt][c][i] = z;
          float hv = (float)h_tile[m * HD + o];
          rh_tile[m * HD + o] = (__bf16)(r * hv);
        }
      }
    __syncthreads();

    // ---- candidate + hidden update ----
    f32x8 pac[2][4];
#pragma unroll
    for (int rt = 0; rt < 2; ++rt)
#pragma unroll
      for (int c = 0; c < 4; ++c) pac[rt][c] = fzero8();
    for (int ks = 0; ks < 16; ++ks) {
      bf16x16 a0 = frag_a(&rh_tile[(size_t)lc * HD + ks * 32 + lh * 8]);
      bf16x16 a1 = frag_a(&rh_tile[(size_t)(16 + lc) * HD + ks * 32 + lh * 8]);
#pragma unroll
      for (int c = 0; c < 4; ++c) {
        int o = cb + c * 16 + lc;
        bf16x16 bhf = frag_b(&Whnb[(size_t)o * HD + ks * 32 + lh * 8]);
        pac[0][c] = wmma_bf16(a0, bhf, pac[0][c]);
        pac[1][c] = wmma_bf16(a1, bhf, pac[1][c]);
      }
    }
#pragma unroll
    for (int rt = 0; rt < 2; ++rt)
#pragma unroll
      for (int c = 0; c < 4; ++c) {
        int o = cb + c * 16 + lc;
#pragma unroll
        for (int i = 0; i < 8; ++i) {
          int m = rt * 16 + i + 8 * lh;
          float pre = tanhf(pac[rt][c][i] + Whnsel[o * 5 + xidx[rt][i]] + bh[o]);
          float z = zac[rt][c][i];
          float hv = (float)h_tile[m * HD + o];
          float nh = (1.0f - z) * hv + z * pre;
          h_tile[m * HD + o] = (__bf16)nh;
        }
      }
    __syncthreads();

    // ---- fused nuc head hidden ----
    f32x8 hac[2][4];
#pragma unroll
    for (int rt = 0; rt < 2; ++rt)
#pragma unroll
      for (int c = 0; c < 4; ++c) hac[rt][c] = fzero8();
    for (int ks = 0; ks < 16; ++ks) {
      bf16x16 a0 = frag_a(&h_tile[(size_t)lc * HD + ks * 32 + lh * 8]);
      bf16x16 a1 = frag_a(&h_tile[(size_t)(16 + lc) * HD + ks * 32 + lh * 8]);
#pragma unroll
      for (int c = 0; c < 4; ++c) {
        int o = cb + c * 16 + lc;
        bf16x16 bnf = frag_b(&Wnlb[(size_t)o * HD + ks * 32 + lh * 8]);
        hac[0][c] = wmma_bf16(a0, bnf, hac[0][c]);
        hac[1][c] = wmma_bf16(a1, bnf, hac[1][c]);
      }
    }
#pragma unroll
    for (int rt = 0; rt < 2; ++rt)
#pragma unroll
      for (int c = 0; c < 4; ++c) {
        int o = cb + c * 16 + lc;
#pragma unroll
        for (int i = 0; i < 8; ++i) {
          int m = rt * 16 + i + 8 * lh;
          float v = hac[rt][c][i] + lat_lds[m * HD + o];
          hid_tile[m * HD + o] = (__bf16)(v > 0.f ? v : 0.f);
        }
      }
    __syncthreads();

    // ---- logits as padded 16x16 WMMA tiles (waves 0,1 handle row-tiles 0,1) ----
    if (wv < 2) {
      const int rt = wv;
      f32x8 cfr = fzero8();
      for (int ks = 0; ks < 16; ++ks)
        cfr = wmma_bf16(frag_a(&hid_tile[(size_t)(rt * 16 + lc) * HD + ks * 32 + lh * 8]),
                        frag_b(&Wnucb[(size_t)lc * HD + ks * 32 + lh * 8]), cfr);
      if (lc < 5) {
#pragma unroll
        for (int i = 0; i < 8; ++i) {
          int m = rt * 16 + i + 8 * lh;
          out[NUC_OFF + ((size_t)(row0 + m) * LSEQ + t) * 5 + lc] = cfr[i] + b_nuc[lc];
        }
      }
    }
  }
  __syncthreads();

  for (int i = tid; i < 32 * HD; i += 256) out[(size_t)row0 * HD + i] = (float)h_tile[i];

  // ---- hpn head ----
  f32x8 hac[2][4];
#pragma unroll
  for (int rt = 0; rt < 2; ++rt)
#pragma unroll
    for (int c = 0; c < 4; ++c) hac[rt][c] = fzero8();
  for (int ks = 0; ks < 16; ++ks) {
    bf16x16 a0 = frag_a(&h_tile[(size_t)lc * HD + ks * 32 + lh * 8]);
    bf16x16 a1 = frag_a(&h_tile[(size_t)(16 + lc) * HD + ks * 32 + lh * 8]);
#pragma unroll
    for (int c = 0; c < 4; ++c) {
      int o = cb + c * 16 + lc;
      bf16x16 bpf = frag_b(&Whpnlb[(size_t)o * HD + ks * 32 + lh * 8]);
      hac[0][c] = wmma_bf16(a0, bpf, hac[0][c]);
      hac[1][c] = wmma_bf16(a1, bpf, hac[1][c]);
    }
  }
#pragma unroll
  for (int rt = 0; rt < 2; ++rt)
#pragma unroll
    for (int c = 0; c < 4; ++c) {
      int o = cb + c * 16 + lc;
#pragma unroll
      for (int i = 0; i < 8; ++i) {
        int m = rt * 16 + i + 8 * lh;
        float v = hac[rt][c][i] + b_hpn_nl[o];
        hid_tile[m * HD + o] = (__bf16)(v > 0.f ? v : 0.f);
      }
    }
  __syncthreads();
  if (wv < 2) {
    const int rt = wv;
    f32x8 cfr = fzero8();
    for (int ks = 0; ks < 16; ++ks)
      cfr = wmma_bf16(frag_a(&hid_tile[(size_t)(rt * 16 + lc) * HD + ks * 32 + lh * 8]),
                      frag_b(&Whpnb[(size_t)lc * HD + ks * 32 + lh * 8]), cfr);
    if (lc < 4) {
#pragma unroll
      for (int i = 0; i < 8; ++i) {
        int m = rt * 16 + i + 8 * lh;
        out[HPN_OFF + (size_t)(row0 + m) * 4 + lc] = cfr[i] + b_hpn[lc];
      }
    }
  }
}

// ---------------- host ----------------
extern "C" void kernel_launch(void* const* d_in, const int* in_sizes, int n_in,
                              void* d_out, int out_size, void* d_ws, size_t ws_size,
                              hipStream_t stream) {
  const int*   hpn_idx   = (const int*)d_in[0];
  const int*   nuc_idx   = (const int*)d_in[1];
  const float* h_nei     = (const float*)d_in[2];
  const float* glat      = (const float*)d_in[3];
  const float* Wz_mp     = (const float*)d_in[4];
  const float* bz_mp     = (const float*)d_in[5];
  const float* Wr_mp     = (const float*)d_in[6];
  const float* br_mp     = (const float*)d_in[7];
  const float* Ur_mp     = (const float*)d_in[8];
  const float* Wh_mp     = (const float*)d_in[9];
  const float* bh_mp     = (const float*)d_in[10];
  const float* Wz_nuc    = (const float*)d_in[11];
  const float* bz_nuc    = (const float*)d_in[12];
  const float* Wr_nuc    = (const float*)d_in[13];
  const float* br_nuc    = (const float*)d_in[14];
  const float* Wh_nuc    = (const float*)d_in[15];
  const float* bh_nuc    = (const float*)d_in[16];
  const float* W_hpn_nl  = (const float*)d_in[17];
  const float* b_hpn_nl  = (const float*)d_in[18];
  const float* W_hpn     = (const float*)d_in[19];
  const float* b_hpn     = (const float*)d_in[20];
  const float* W_nuc_nl  = (const float*)d_in[21];
  const float* b_nuc_nl  = (const float*)d_in[22];
  const float* W_nuc     = (const float*)d_in[23];
  const float* b_nuc     = (const float*)d_in[24];
  const float* W_topo_nl = (const float*)d_in[25];
  const float* b_topo_nl = (const float*)d_in[26];
  const float* W_topo    = (const float*)d_in[27];
  const float* b_topo    = (const float*)d_in[28];

  char* base = (char*)d_ws;
  size_t off = 0;
  auto take = [&](size_t bytes) -> void* {
    void* p = base + off;
    off += (bytes + 255) & ~size_t(255);
    return p;
  };
  __bf16* Wzb     = (__bf16*)take(262144 * 2);
  __bf16* Whb     = (__bf16*)take(262144 * 2);
  __bf16* Urb     = (__bf16*)take(262144 * 2);
  __bf16* Wznb    = (__bf16*)take(262144 * 2);
  __bf16* Wrnb    = (__bf16*)take(262144 * 2);
  __bf16* Whnb    = (__bf16*)take(262144 * 2);
  __bf16* Wnlb    = (__bf16*)take(262144 * 2);
  __bf16* Wlatb   = (__bf16*)take(65536 * 2);
  __bf16* Whpnlb  = (__bf16*)take(262144 * 2);
  __bf16* Wtopolb = (__bf16*)take(262144 * 2);
  __bf16* Glatb   = (__bf16*)take(262144 * 2);
  __bf16* Wnucb   = (__bf16*)take(8192 * 2);
  __bf16* Whpnb   = (__bf16*)take(8192 * 2);
  __bf16* Wtopob  = (__bf16*)take(8192 * 2);
  float* Wzsel   = (float*)take(2048 * 4);
  float* Whsel   = (float*)take(2048 * 4);
  float* Wznsel  = (float*)take(2560 * 4);
  float* Wrnsel  = (float*)take(2560 * 4);
  float* Whnsel  = (float*)take(2560 * 4);
  float* msgbuf  = (float*)take((size_t)NTOT * HD * 4);
  float* latbuf  = (float*)take((size_t)NTOT * HD * 4);
  (void)ws_size; (void)in_sizes; (void)n_in; (void)out_size;

  float* out = (float*)d_out;
  const int G512 = (512 * 512 + 255) / 256;
  const int G128 = (512 * 128 + 255) / 256;
  const int Gpad = (16 * 512 + 255) / 256;

  cvt_slice<<<G512, 256, 0, stream>>>(Wz_mp,    Wzb,    512, 512, 516, 4, 512);
  cvt_slice<<<G512, 256, 0, stream>>>(Wh_mp,    Whb,    512, 512, 516, 4, 512);
  cvt_slice<<<G512, 256, 0, stream>>>(Ur_mp,    Urb,    512, 512, 512, 0, 512);
  cvt_slice<<<G512, 256, 0, stream>>>(Wz_nuc,   Wznb,   512, 512, 517, 5, 512);
  cvt_slice<<<G512, 256, 0, stream>>>(Wr_nuc,   Wrnb,   512, 512, 517, 5, 512);
  cvt_slice<<<G512, 256, 0, stream>>>(Wh_nuc,   Whnb,   512, 512, 517, 5, 512);
  cvt_slice<<<G512, 256, 0, stream>>>(W_nuc_nl, Wnlb,   512, 512, 640, 0, 512);
  cvt_slice<<<G128, 256, 0, stream>>>(W_nuc_nl, Wlatb,  512, 128, 640, 512, 512);
  cvt_slice<<<G512, 256, 0, stream>>>(W_hpn_nl, Whpnlb, 512, 512, 512, 0, 512);
  cvt_slice<<<G512, 256, 0, stream>>>(W_topo_nl,Wtopolb,512, 512, 512, 0, 512);
  cvt_slice<<<G512, 256, 0, stream>>>(glat,     Glatb,  2048, 128, 128, 0, 2048);
  cvt_slice<<<Gpad, 256, 0, stream>>>(W_nuc,    Wnucb,  5,   512, 512, 0, 16);
  cvt_slice<<<Gpad, 256, 0, stream>>>(W_hpn,    Whpnb,  4,   512, 512, 0, 16);
  cvt_slice<<<Gpad, 256, 0, stream>>>(W_topo,   Wtopob, 1,   512, 512, 0, 16);
  copy_slice<<<(2048 + 255) / 256, 256, 0, stream>>>(Wz_mp,  Wzsel,  512, 4, 516, 0);
  copy_slice<<<(2048 + 255) / 256, 256, 0, stream>>>(Wh_mp,  Whsel,  512, 4, 516, 0);
  copy_slice<<<(2560 + 255) / 256, 256, 0, stream>>>(Wz_nuc, Wznsel, 512, 5, 517, 0);
  copy_slice<<<(2560 + 255) / 256, 256, 0, stream>>>(Wr_nuc, Wrnsel, 512, 5, 517, 0);
  copy_slice<<<(2560 + 255) / 256, 256, 0, stream>>>(Wh_nuc, Whnsel, 512, 5, 517, 0);

  lat_kernel<<<NTOT / 16, 256, 0, stream>>>(Glatb, Wlatb, b_nuc_nl, latbuf);

  size_t msg_smem = 131072 + 32768 + 16384 + 16384 + 16384;   // 208 KB
  msg_kernel<<<NTOT / 16, 256, msg_smem, stream>>>(
      hpn_idx, h_nei, Wzb, Wzsel, bz_mp, Wr_mp, br_mp, Urb, Whb, Whsel, bh_mp,
      Wtopolb, b_topo_nl, Wtopob, b_topo, msgbuf, out);

  size_t scan_smem = 32768 + 32768 + 32768 + 65536;           // 160 KB
  scan_kernel<<<NTOT / 32, 256, scan_smem, stream>>>(
      nuc_idx, msgbuf, latbuf, Wznb, Wrnb, Whnb, Wznsel, Wrnsel, Whnsel,
      bz_nuc, br_nuc, bh_nuc, Wnlb, Wnucb, b_nuc, Whpnlb, b_hpn_nl, Whpnb, b_hpn, out);
}
